// LinformerMultiHeadSelfAttention_13280038879464
// MI455X (gfx1250) — matrix operand order repaired and checked by reading, not compiled, not run on hardware
//
#include <hip/hip_runtime.h>

// ---------------------------------------------------------------------------
// Linformer MHSA for MI455X (gfx1250, wave32, WMMA bf16 -> f32 accum)
//   B=4, N=4096, F=1024, H=16, d=64, E=256
// Stages:
//   0) x/weights/proj fp32 -> bf16 (+ transpose proj to [E][N])
//   1) Q/K/V = x @ W^T                  (WMMA GEMM, async double-buffered LDS)
//   2) Kp/Vp = projT @ K|V  per batch   (WMMA GEMM, B in [K][N] layout)
//   3) fused scores/softmax/PV per (b,h,token-tile)  (WMMA + shfl softmax)
//   4) out = ctx @ Wo^T  (f32 output)
// Data movement uses GLOBAL_LOAD_ASYNC_TO_LDS_B128 + s_wait_asynccnt (CDNA5
// async path, ASYNCcnt-tracked), bypassing VGPR staging.
// ---------------------------------------------------------------------------

#define SEQ_LEN  4096
#define FEAT_DIM 1024
#define NUM_HEAD 16
#define HEAD_DIM 64
#define PROJ_DIM 256
#define BATCH    4

typedef __bf16          v16bf __attribute__((ext_vector_type(16)));
typedef float           v8f   __attribute__((ext_vector_type(8)));
typedef unsigned short  v8u   __attribute__((ext_vector_type(8)));
typedef unsigned short  v16u  __attribute__((ext_vector_type(16)));

static __device__ __forceinline__ unsigned short f2bf(float f) {
  unsigned int u = __float_as_uint(f);
  return (unsigned short)((u + 0x7FFFu + ((u >> 16) & 1u)) >> 16);  // RNE
}

static __device__ __forceinline__ v16bf mk16(v8u lo, v8u hi) {
  v16u t;
#pragma unroll
  for (int i = 0; i < 8; ++i) { t[i] = lo[i]; t[i + 8] = hi[i]; }
  return __builtin_bit_cast(v16bf, t);
}

static __device__ __forceinline__ v8f wmma_bf16(v16bf a, v16bf b, v8f c) {
  // D = A(16x32 bf16) * B(32x16 bf16) + C(16x16 f32)
  return __builtin_amdgcn_wmma_f32_16x16x32_bf16(false, a, false, b,
                                                 (short)0, c, false, false);
}

// ---- CDNA5 async global->LDS (ASYNCcnt) -----------------------------------
// Copies 16B per lane: LDS[vdst] = MEM[saddr + vaddr]   (GVS mode, offset 0)
static __device__ __forceinline__ unsigned lds_off(const void* p) {
  return (unsigned)(unsigned long)p;  // low 32 bits of generic ptr = LDS addr
}
static __device__ __forceinline__ void async_load_b128(
    const unsigned short* base, unsigned voff, unsigned ldsoff) {
  asm volatile("global_load_async_to_lds_b128 %0, %1, %2"
               :
               : "v"(ldsoff), "v"(voff), "s"(base)
               : "memory");
}
static __device__ __forceinline__ void wait_async0() {
  asm volatile("s_wait_asynccnt 0x0" ::: "memory");
}
static __device__ __forceinline__ void wait_async4() {
  asm volatile("s_wait_asynccnt 0x4" ::: "memory");
}

// --------------------------- prep kernels ----------------------------------

__global__ void cvt_f32_bf16(const float* __restrict__ in,
                             unsigned short* __restrict__ out, long n) {
  long i = (long)blockIdx.x * blockDim.x + threadIdx.x;
  if (i < n) out[i] = f2bf(in[i]);
}

// in: [R][C] fp32  ->  out: [C][R] bf16
__global__ void transpose_cvt(const float* __restrict__ in,
                              unsigned short* __restrict__ out, int R, int C) {
  long i = (long)blockIdx.x * blockDim.x + threadIdx.x;
  if (i < (long)R * C) {
    int r = (int)(i / C), c = (int)(i % C);
    out[(long)c * R + r] = f2bf(in[i]);
  }
}

// --------------------------- generic WMMA GEMM -----------------------------
// C[M][N] = A[M][K] * (BT ? Bp[N][K]^T : Bp[K][N])
// Block tile 128x128, 8 waves of 64x32, K-step 32.
// Double-buffered LDS, async global->LDS copies, s_wait_asynccnt fencing.
// Per wave per tile: 2 A-chunks + 2 B-chunks = 4 async instrs -> wait<=4
// keeps next tile's loads in flight while guaranteeing current tile landed.

template <bool BT, bool C_F32>
__global__ __launch_bounds__(256, 1) void gemm_wmma(
    const unsigned short* __restrict__ Ap, const unsigned short* __restrict__ Bp,
    void* __restrict__ Cp, int M, int N, int K, long sA, long sB, long sC) {
  __shared__ unsigned short As[2][128][40];                       // [m][k]+pad
  __shared__ unsigned short Bs[2][BT ? 128 : 32][BT ? 40 : 136]; // [n][k]/[k][n]

  const int tid  = threadIdx.x;
  const int lane = tid & 31;
  const int wave = tid >> 5;
  const int wm = wave >> 2;            // 0..1  -> M offset wm*64
  const int wn = wave & 3;             // 0..3  -> N offset wn*32
  const int m0 = blockIdx.y * 128;
  const int n0 = blockIdx.x * 128;
  const int zb = blockIdx.z;
  (void)M;

  const unsigned short* Au = Ap + (long)zb * sA;
  const unsigned short* Bb = Bp + (long)zb * sB;

  auto stage = [&](int p, int kk) {
#pragma unroll
    for (int it = 0; it < 2; ++it) {           // A tile 128x32
      const int idx = tid + it * 256;
      const int r = idx >> 2, c = (idx & 3) * 8;
      const unsigned voff = (unsigned)(((long)(m0 + r) * K + kk + c) * 2);
      async_load_b128(Au, voff, lds_off(&As[p][r][c]));
    }
    if constexpr (BT) {                        // Bp[N][K] rows = out columns
#pragma unroll
      for (int it = 0; it < 2; ++it) {
        const int idx = tid + it * 256;
        const int r = idx >> 2, c = (idx & 3) * 8;
        const unsigned voff = (unsigned)(((long)(n0 + r) * K + kk + c) * 2);
        async_load_b128(Bb, voff, lds_off(&Bs[p][r][c]));
      }
    } else {                                   // Bp[K][N] row-major
#pragma unroll
      for (int it = 0; it < 2; ++it) {
        const int idx = tid + it * 256;
        const int r = idx >> 4, c = (idx & 15) * 8;
        const unsigned voff = (unsigned)(((long)(kk + r) * N + n0 + c) * 2);
        async_load_b128(Bb, voff, lds_off(&Bs[p][r][c]));
      }
    }
  };

  v8f acc[4][2] = {};

  stage(0, 0);
  int p = 0;
  for (int kk = 0; kk < K; kk += 32, p ^= 1) {
    const bool more = (kk + 32) < K;
    if (more) stage(p ^ 1, kk + 32);   // prefetch next tile (other buffer)
    if (more) wait_async4(); else wait_async0();  // current tile landed
    __syncthreads();                   // all waves' copies visible

    const int fl = lane & 15;
    const int ka = (lane < 16) ? 0 : 16;  // A frag: K base (contiguous 16)
    const int kb = (lane < 16) ? 0 : 16;  // B frag: K row base

    v16bf bf[2];
#pragma unroll
    for (int nt = 0; nt < 2; ++nt) {
      const int n = wn * 32 + nt * 16 + fl;
      if constexpr (BT) {
        bf[nt] = mk16(*(const v8u*)&Bs[p][n][kb], *(const v8u*)&Bs[p][n][kb + 8]);
      } else {
        v16u t;
#pragma unroll
        for (int i = 0; i < 16; ++i) t[i] = Bs[p][kb + i][n];
        bf[nt] = __builtin_bit_cast(v16bf, t);
      }
    }
#pragma unroll
    for (int mt = 0; mt < 4; ++mt) {
      const int m = wm * 64 + mt * 16 + fl;
      v16bf af = mk16(*(const v8u*)&As[p][m][ka], *(const v8u*)&As[p][m][ka + 8]);
#pragma unroll
      for (int nt = 0; nt < 2; ++nt)
        acc[mt][nt] = wmma_bf16(af, bf[nt], acc[mt][nt]);
    }
    __syncthreads();                   // reads done before buffer p is reused
  }

  // ---- epilogue: C layout lane l holds col=l&15, rows r+(l<16?0:8) ----
  const int fl = lane & 15;
  const int rb = (lane < 16) ? 0 : 8;
#pragma unroll
  for (int mt = 0; mt < 4; ++mt)
#pragma unroll
    for (int nt = 0; nt < 2; ++nt)
#pragma unroll
      for (int r = 0; r < 8; ++r) {
        const long row = m0 + wm * 64 + mt * 16 + rb + r;
        const long col = n0 + wn * 32 + nt * 16 + fl;
        const float v = acc[mt][nt][r];
        if constexpr (C_F32)
          ((float*)Cp)[(long)zb * sC + row * (long)N + col] = v;
        else
          ((unsigned short*)Cp)[(long)zb * sC + row * (long)N + col] = f2bf(v);
      }
}

// --------------------------- fused attention -------------------------------
// grid (N/128, H, B), block 256 (8 waves, 16 tokens each).
// Q[B,N,H,64], Kp/Vp[B,256,1024] (head-major columns), ctx[B,N,H,64] bf16.

__global__ __launch_bounds__(256, 1) void attn_kernel(
    const unsigned short* __restrict__ Q, const unsigned short* __restrict__ Kp,
    const unsigned short* __restrict__ Vp, unsigned short* __restrict__ ctx) {
  __shared__ unsigned short Qs[128][72];       // [token][d]
  __shared__ unsigned short Ks[PROJ_DIM][72];  // [e][d]
  __shared__ unsigned short Vs[PROJ_DIM][72];  // [e][d]
  __shared__ unsigned short Ps[8][16][264];    // per-wave P (16 tok x 256 e)

  const int tid  = threadIdx.x;
  const int lane = tid & 31;
  const int wave = tid >> 5;
  const int tb = blockIdx.x, h = blockIdx.y, bb = blockIdx.z;

  const unsigned short* Qg = Q  + ((long)bb * SEQ_LEN + (long)tb * 128) * FEAT_DIM + h * HEAD_DIM;
  const unsigned short* Kg = Kp + (long)bb * PROJ_DIM * FEAT_DIM + h * HEAD_DIM;
  const unsigned short* Vg = Vp + (long)bb * PROJ_DIM * FEAT_DIM + h * HEAD_DIM;

  // async stage Q (128x64) and Kp/Vp (256x64) tiles straight into LDS
#pragma unroll
  for (int it = 0; it < 4; ++it) {
    const int idx = tid + it * 256;            // 1024 16B chunks
    const int r = idx >> 3, c = (idx & 7) * 8;
    async_load_b128(Qg, (unsigned)(((long)r * FEAT_DIM + c) * 2),
                    lds_off(&Qs[r][c]));
  }
#pragma unroll
  for (int it = 0; it < 8; ++it) {
    const int idx = tid + it * 256;            // 2048 16B chunks each
    const int r = idx >> 3, c = (idx & 7) * 8;
    const unsigned voff = (unsigned)(((long)r * FEAT_DIM + c) * 2);
    async_load_b128(Kg, voff, lds_off(&Ks[r][c]));
    async_load_b128(Vg, voff, lds_off(&Vs[r][c]));
  }
  wait_async0();
  __syncthreads();

  const int fl = lane & 15;
  const int ka = (lane < 16) ? 0 : 16;
  const int kb = (lane < 16) ? 0 : 16;
  const int rb = (lane < 16) ? 0 : 8;
  const int mrow = wave * 16;

  // ---- scores S = Q Kp^T : K-dim = d = 64 (2 steps), 16 e-tiles ----
  v16bf aq[2];
#pragma unroll
  for (int kt = 0; kt < 2; ++kt)
    aq[kt] = mk16(*(const v8u*)&Qs[mrow + fl][kt * 32 + ka],
                  *(const v8u*)&Qs[mrow + fl][kt * 32 + ka + 8]);

  v8f s[16] = {};
#pragma unroll
  for (int t = 0; t < 16; ++t)
#pragma unroll
    for (int kt = 0; kt < 2; ++kt) {
      v16bf bk = mk16(*(const v8u*)&Ks[t * 16 + fl][kt * 32 + kb],
                      *(const v8u*)&Ks[t * 16 + fl][kt * 32 + kb + 8]);
      s[t] = wmma_bf16(aq[kt], bk, s[t]);
    }

  // ---- softmax over e (scale 1/8 folded into exp2) ----
  const float c0 = 1.44269504f * 0.125f;  // log2(e)/tau
  float inv[8];
#pragma unroll
  for (int r = 0; r < 8; ++r) {
    float mx = -3.4e38f;
#pragma unroll
    for (int t = 0; t < 16; ++t) mx = fmaxf(mx, s[t][r]);
    mx = fmaxf(mx, __shfl_xor(mx, 1, 32));
    mx = fmaxf(mx, __shfl_xor(mx, 2, 32));
    mx = fmaxf(mx, __shfl_xor(mx, 4, 32));
    mx = fmaxf(mx, __shfl_xor(mx, 8, 32));
    float sum = 0.f;
#pragma unroll
    for (int t = 0; t < 16; ++t) {
      float pv = exp2f((s[t][r] - mx) * c0);
      s[t][r] = pv;
      sum += pv;
    }
    sum += __shfl_xor(sum, 1, 32);
    sum += __shfl_xor(sum, 2, 32);
    sum += __shfl_xor(sum, 4, 32);
    sum += __shfl_xor(sum, 8, 32);
    inv[r] = 1.0f / sum;
  }

  // ---- P to LDS (re-layout C-frag -> A-frag), unnormalized ----
#pragma unroll
  for (int t = 0; t < 16; ++t)
#pragma unroll
    for (int r = 0; r < 8; ++r)
      Ps[wave][rb + r][t * 16 + fl] = f2bf(s[t][r]);
  __syncthreads();

  // ---- O = P Vp : K-dim = e = 256 (8 steps), 4 d-tiles ----
  v8f o[4] = {};
#pragma unroll
  for (int ks = 0; ks < 8; ++ks) {
    v16bf ap = mk16(*(const v8u*)&Ps[wave][fl][ks * 32 + ka],
                    *(const v8u*)&Ps[wave][fl][ks * 32 + ka + 8]);
#pragma unroll
    for (int nt = 0; nt < 4; ++nt) {
      v16u tv;
#pragma unroll
      for (int i = 0; i < 16; ++i) tv[i] = Vs[ks * 32 + kb + i][nt * 16 + fl];
      o[nt] = wmma_bf16(ap, __builtin_bit_cast(v16bf, tv), o[nt]);
    }
  }

  // ---- store ctx (apply deferred 1/sum) ----
#pragma unroll
  for (int nt = 0; nt < 4; ++nt)
#pragma unroll
    for (int r = 0; r < 8; ++r) {
      const long row = (long)tb * 128 + mrow + rb + r;
      const long col = h * HEAD_DIM + nt * 16 + fl;
      ctx[((long)bb * SEQ_LEN + row) * FEAT_DIM + col] = f2bf(o[nt][r] * inv[r]);
    }
}

// --------------------------- host launcher ---------------------------------

extern "C" void kernel_launch(void* const* d_in, const int* in_sizes, int n_in,
                              void* d_out, int out_size, void* d_ws,
                              size_t ws_size, hipStream_t stream) {
  (void)in_sizes; (void)n_in; (void)out_size; (void)ws_size;

  const float* x  = (const float*)d_in[0];
  const float* wq = (const float*)d_in[1];
  const float* wk = (const float*)d_in[2];
  const float* wv = (const float*)d_in[3];
  const float* wo = (const float*)d_in[4];
  const float* pe = (const float*)d_in[5];
  const float* pf = (const float*)d_in[6];

  char* ws = (char*)d_ws;
  size_t off = 0;
  auto carve = [&](size_t bytes) {
    void* p = ws + off;
    off = (off + bytes + 255) & ~(size_t)255;
    return p;
  };
  const long MQ = (long)BATCH * SEQ_LEN;  // 16384 token rows

  unsigned short* xb   = (unsigned short*)carve((size_t)MQ * FEAT_DIM * 2);
  unsigned short* wq_b = (unsigned short*)carve((size_t)FEAT_DIM * FEAT_DIM * 2);
  unsigned short* wk_b = (unsigned short*)carve((size_t)FEAT_DIM * FEAT_DIM * 2);
  unsigned short* wv_b = (unsigned short*)carve((size_t)FEAT_DIM * FEAT_DIM * 2);
  unsigned short* wo_b = (unsigned short*)carve((size_t)FEAT_DIM * FEAT_DIM * 2);
  unsigned short* peT  = (unsigned short*)carve((size_t)PROJ_DIM * SEQ_LEN * 2);
  unsigned short* pfT  = (unsigned short*)carve((size_t)PROJ_DIM * SEQ_LEN * 2);
  unsigned short* Qb   = (unsigned short*)carve((size_t)MQ * FEAT_DIM * 2);
  unsigned short* Kb   = (unsigned short*)carve((size_t)MQ * FEAT_DIM * 2);
  unsigned short* Vb   = (unsigned short*)carve((size_t)MQ * FEAT_DIM * 2);
  unsigned short* Kpj  = (unsigned short*)carve((size_t)BATCH * PROJ_DIM * FEAT_DIM * 2);
  unsigned short* Vpj  = (unsigned short*)carve((size_t)BATCH * PROJ_DIM * FEAT_DIM * 2);
  unsigned short* ctx  = (unsigned short*)carve((size_t)MQ * FEAT_DIM * 2);

  // 0) precision prep
  const long nX = MQ * FEAT_DIM;
  cvt_f32_bf16<<<(unsigned)((nX + 255) / 256), 256, 0, stream>>>(x, xb, nX);
  const long nW = (long)FEAT_DIM * FEAT_DIM;
  cvt_f32_bf16<<<(unsigned)((nW + 255) / 256), 256, 0, stream>>>(wq, wq_b, nW);
  cvt_f32_bf16<<<(unsigned)((nW + 255) / 256), 256, 0, stream>>>(wk, wk_b, nW);
  cvt_f32_bf16<<<(unsigned)((nW + 255) / 256), 256, 0, stream>>>(wv, wv_b, nW);
  cvt_f32_bf16<<<(unsigned)((nW + 255) / 256), 256, 0, stream>>>(wo, wo_b, nW);
  const long nP = (long)SEQ_LEN * PROJ_DIM;
  transpose_cvt<<<(unsigned)((nP + 255) / 256), 256, 0, stream>>>(pe, peT, SEQ_LEN, PROJ_DIM);
  transpose_cvt<<<(unsigned)((nP + 255) / 256), 256, 0, stream>>>(pf, pfT, SEQ_LEN, PROJ_DIM);

  // 1) Q/K/V = x @ W^T   (M=16384, N=1024, K=1024)
  dim3 g1(FEAT_DIM / 128, MQ / 128, 1);
  gemm_wmma<true, false><<<g1, 256, 0, stream>>>(
      xb, wq_b, Qb, (int)MQ, FEAT_DIM, FEAT_DIM, 0, 0, 0);
  gemm_wmma<true, false><<<g1, 256, 0, stream>>>(
      xb, wk_b, Kb, (int)MQ, FEAT_DIM, FEAT_DIM, 0, 0, 0);
  gemm_wmma<true, false><<<g1, 256, 0, stream>>>(
      xb, wv_b, Vb, (int)MQ, FEAT_DIM, FEAT_DIM, 0, 0, 0);

  // 2) Kp/Vp = projT @ K|V  per batch  (M=256, N=1024, K=4096)
  dim3 g2(FEAT_DIM / 128, PROJ_DIM / 128, BATCH);
  gemm_wmma<false, false><<<g2, 256, 0, stream>>>(
      peT, Kb, Kpj, PROJ_DIM, FEAT_DIM, SEQ_LEN, 0,
      (long)SEQ_LEN * FEAT_DIM, (long)PROJ_DIM * FEAT_DIM);
  gemm_wmma<false, false><<<g2, 256, 0, stream>>>(
      pfT, Vb, Vpj, PROJ_DIM, FEAT_DIM, SEQ_LEN, 0,
      (long)SEQ_LEN * FEAT_DIM, (long)PROJ_DIM * FEAT_DIM);

  // 3) fused attention
  dim3 g3(SEQ_LEN / 128, NUM_HEAD, BATCH);
  attn_kernel<<<g3, 256, 0, stream>>>(Qb, Kpj, Vpj, ctx);

  // 4) out = ctx @ Wo^T  (f32 output)
  gemm_wmma<true, true><<<g1, 256, 0, stream>>>(
      ctx, wo_b, d_out, (int)MQ, FEAT_DIM, FEAT_DIM, 0, 0, 0);
}
